// EnhancedContextAwareDualVQ_24902220382739
// MI455X (gfx1250) — matrix operand order repaired and checked by reading, not compiled, not run on hardware
//
#include <hip/hip_runtime.h>
#include <hip/hip_bf16.h>

// ---------------------------------------------------------------------------
// CDNA5 (gfx1250) dual-VQ forward, GEMMs on v_wmma_f32_16x16x32_f16 with
// double-buffered LDS staging (global loads for tile k+1 overlap WMMA on k).
// ---------------------------------------------------------------------------

typedef __attribute__((ext_vector_type(8)))  float     v8f;
typedef __attribute__((ext_vector_type(16))) _Float16  v16h;
typedef __attribute__((ext_vector_type(8)))  _Float16  h8;
typedef __attribute__((ext_vector_type(4)))  _Float16  h4;

constexpr int EPI_BIAS   = 0;  // f32 out: acc + bias[n]
constexpr int EPI_RELU16 = 1;  // f16 out: relu(acc + bias[n])
constexpr int EPI_DIST   = 2;  // f32 out: bias[n] - 2*acc  (bias = ||cb row||^2)

#define BMh 64
#define BNh 128
#define BKh 32
#define LDSH 40          // row stride in halfs (32 + 8 pad, keeps 16B alignment)
#define GEMM_THREADS 128

__device__ __forceinline__ v16h pack16(const _Float16* p0, const _Float16* p1)
{
    h8 a = *(const h8*)p0;
    h8 b = *(const h8*)p1;
    v16h r;
#pragma unroll
    for (int i = 0; i < 8; i++) { r[i] = a[i]; r[i + 8] = b[i]; }
    return r;
}

// ---------------------------------------------------------------------------
// C[M x N] = A[M x K](f16,row-major) * Bt[N x K](f16,row-major)^T + epilogue
// 128 threads = 4 waves (2x2); each wave computes 32x64 (2x4 WMMA frags).
// Double-buffered LDS: fetch tile k+1 while computing tile k.
// ---------------------------------------------------------------------------
template<int EPI>
__global__ __launch_bounds__(GEMM_THREADS)
void gemm_wmma_f16(const _Float16* __restrict__ A, const _Float16* __restrict__ Bt,
                   const float* __restrict__ bias, void* __restrict__ Cv,
                   int M, int N, int K)
{
    __shared__ _Float16 As[2][BMh][LDSH];
    __shared__ _Float16 Bs[2][BNh][LDSH];

    const int tid    = threadIdx.x;
    const int lane   = tid & 31;
    const int wave   = tid >> 5;            // 0..3
    const int wm     = (wave >> 1) * 32;    // wave row offset in block tile
    const int wn     = (wave & 1) * 64;     // wave col offset
    const int m0     = blockIdx.y * BMh;
    const int n0     = blockIdx.x * BNh;
    const int l15    = lane & 15;
    const int ka     = (lane >> 4) * 8;     // A: K chunks {ka..ka+7, ka+16..ka+23}
    const int kb     = (lane >> 4) * 8;     // B: K chunks {kb..kb+7, kb+16..kb+23}
    const int rhalf  = (lane >> 4) * 8;     // C/D: M = r (+8 for high lanes)

    // per-thread staging coordinates (16B chunks)
    int rA[2], cA[2], rB[4], cB[4];
    const _Float16* pA[2];
    const _Float16* pB[4];
#pragma unroll
    for (int i = 0; i < 2; i++) {
        int idx = tid + i * GEMM_THREADS;
        rA[i] = idx >> 2; cA[i] = (idx & 3) * 8;
        pA[i] = A + (size_t)(m0 + rA[i]) * K + cA[i];
    }
#pragma unroll
    for (int i = 0; i < 4; i++) {
        int idx = tid + i * GEMM_THREADS;
        rB[i] = idx >> 2; cB[i] = (idx & 3) * 8;
        pB[i] = Bt + (size_t)(n0 + rB[i]) * K + cB[i];
    }

    v8f acc[2][4];
    const v8f vzero = {0.f, 0.f, 0.f, 0.f, 0.f, 0.f, 0.f, 0.f};
#pragma unroll
    for (int i = 0; i < 2; i++)
#pragma unroll
        for (int j = 0; j < 4; j++) acc[i][j] = vzero;

    // ---- prologue: stage tile 0 into buffer 0 ----
    h8 ra[2], rb[4];
#pragma unroll
    for (int i = 0; i < 2; i++) ra[i] = *(const h8*)(pA[i]);
#pragma unroll
    for (int i = 0; i < 4; i++) rb[i] = *(const h8*)(pB[i]);
#pragma unroll
    for (int i = 0; i < 2; i++) *(h8*)&As[0][rA[i]][cA[i]] = ra[i];
#pragma unroll
    for (int i = 0; i < 4; i++) *(h8*)&Bs[0][rB[i]][cB[i]] = rb[i];
    __syncthreads();

    int cur = 0;
    for (int k0 = 0; k0 < K; k0 += BKh) {
        // issue global loads for tile k+1 (clamped on last iter; no waits here)
        const int kn = (k0 + BKh < K) ? (k0 + BKh) : k0;
#pragma unroll
        for (int i = 0; i < 2; i++) ra[i] = *(const h8*)(pA[i] + kn);
#pragma unroll
        for (int i = 0; i < 4; i++) rb[i] = *(const h8*)(pB[i] + kn);

        // math on current buffer (12 ds_load_b128 + 8 wmma)
        v16h af[2], bf[4];
#pragma unroll
        for (int mi = 0; mi < 2; mi++) {
            const _Float16* row = &As[cur][wm + mi * 16 + l15][0];
            af[mi] = pack16(row + ka, row + ka + 16);
        }
#pragma unroll
        for (int ni = 0; ni < 4; ni++) {
            const _Float16* row = &Bs[cur][wn + ni * 16 + l15][0];
            bf[ni] = pack16(row + kb, row + kb + 16);
        }
#pragma unroll
        for (int mi = 0; mi < 2; mi++)
#pragma unroll
            for (int ni = 0; ni < 4; ni++)
                acc[mi][ni] = __builtin_amdgcn_wmma_f32_16x16x32_f16(
                    false, af[mi], false, bf[ni],
                    (short)0, acc[mi][ni], false, false);

        // stage tile k+1 into the other buffer (single loadcnt wait)
        const int nxt = cur ^ 1;
#pragma unroll
        for (int i = 0; i < 2; i++) *(h8*)&As[nxt][rA[i]][cA[i]] = ra[i];
#pragma unroll
        for (int i = 0; i < 4; i++) *(h8*)&Bs[nxt][rB[i]][cB[i]] = rb[i];
        __syncthreads();
        cur = nxt;
    }

    // ---- epilogue + store ----
#pragma unroll
    for (int mi = 0; mi < 2; mi++) {
#pragma unroll
        for (int ni = 0; ni < 4; ni++) {
            int colg = n0 + wn + ni * 16 + l15;
            float bv = bias[colg];
#pragma unroll
            for (int r = 0; r < 8; r++) {
                int rowg = m0 + wm + mi * 16 + rhalf + r;
                float v = acc[mi][ni][r];
                if (EPI == EPI_DIST) {
                    ((float*)Cv)[(size_t)rowg * N + colg] = bv - 2.0f * v;
                } else if (EPI == EPI_RELU16) {
                    ((_Float16*)Cv)[(size_t)rowg * N + colg] =
                        (_Float16)fmaxf(v + bv, 0.f);
                } else {
                    ((float*)Cv)[(size_t)rowg * N + colg] = v + bv;
                }
            }
        }
    }
}

// ---------------------------------------------------------------------------
// Block-wide reduction helpers (blockDim.x == 256).
// ---------------------------------------------------------------------------
__device__ __forceinline__ float block_sum(float v, float* red)
{
    int t = threadIdx.x;
    red[t] = v; __syncthreads();
    for (int s = 128; s > 0; s >>= 1) {
        if (t < s) red[t] += red[t + s];
        __syncthreads();
    }
    float r = red[0]; __syncthreads();
    return r;
}

__device__ __forceinline__ float block_max(float v, float* red)
{
    int t = threadIdx.x;
    red[t] = v; __syncthreads();
    for (int s = 128; s > 0; s >>= 1) {
        if (t < s) red[t] = fmaxf(red[t], red[t + s]);
        __syncthreads();
    }
    float r = red[0]; __syncthreads();
    return r;
}

__device__ __forceinline__ void block_argmax(float v, int i, float* red, int* redi,
                                             int& oi)
{
    int t = threadIdx.x;
    red[t] = v; redi[t] = i; __syncthreads();
    for (int s = 128; s > 0; s >>= 1) {
        if (t < s) {
            float v2 = red[t + s]; int i2 = redi[t + s];
            if (v2 > red[t] || (v2 == red[t] && i2 < redi[t])) {
                red[t] = v2; redi[t] = i2;
            }
        }
        __syncthreads();
    }
    oi = redi[0]; __syncthreads();
}

// ---------------------------------------------------------------------------
// LN (no-affine stats) * w + b, relu; width 1024; f32 in -> f16 out.
// ---------------------------------------------------------------------------
__global__ __launch_bounds__(256)
void ln_affine_relu_1024(const float* __restrict__ X, const float* __restrict__ w,
                         const float* __restrict__ b, _Float16* __restrict__ Y)
{
    __shared__ float red[256];
    const int n = blockIdx.x, tid = threadIdx.x;
    const int j = tid * 4;
    float4 v = *(const float4*)(X + (size_t)n * 1024 + j);
    float s  = v.x + v.y + v.z + v.w;
    float s2 = v.x * v.x + v.y * v.y + v.z * v.z + v.w * v.w;
    float S  = block_sum(s, red);
    float S2 = block_sum(s2, red);
    float mean = S * (1.f / 1024.f);
    float inv  = rsqrtf(S2 * (1.f / 1024.f) - mean * mean + 1e-5f);
    h4 o;
    o.x = (_Float16)fmaxf((v.x - mean) * inv * w[j + 0] + b[j + 0], 0.f);
    o.y = (_Float16)fmaxf((v.y - mean) * inv * w[j + 1] + b[j + 1], 0.f);
    o.z = (_Float16)fmaxf((v.z - mean) * inv * w[j + 2] + b[j + 2], 0.f);
    o.w = (_Float16)fmaxf((v.w - mean) * inv * w[j + 3] + b[j + 3], 0.f);
    *(h4*)(Y + (size_t)n * 1024 + j) = o;
}

// ---------------------------------------------------------------------------
// Per-row logit assembly + dual argmax.  One block per row; K-wide rows in LDS.
// ---------------------------------------------------------------------------
template<int K>
__global__ __launch_bounds__(256)
void row_logits(const float* __restrict__ dsqp, const float* __restrict__ ctx,
                const float* __restrict__ adj, const int* __restrict__ prev,
                const float* __restrict__ gate,
                const float* __restrict__ zr, const float* __restrict__ zi,
                int* __restrict__ idx_buf, float* __restrict__ idx_out,
                float* __restrict__ div_out, int Nrows)
{
    __shared__ float sd[K];
    __shared__ float sa[K];
    __shared__ float sc[K];
    __shared__ float red[256];
    __shared__ int   redi[256];

    const int n = blockIdx.x, tid = threadIdx.x;
    const float invK = 1.0f / (float)K;

    float zp = 0.f;
    for (int j = tid; j < 1024; j += 256) {
        float v = (j < 512) ? zr[(size_t)n * 512 + j]
                            : zi[(size_t)n * 512 + (j - 512)];
        zp += v * v;
    }
    float zn = block_sum(zp, red);

    float s1 = 0.f, s2 = 0.f;
    for (int j = tid; j < K; j += 256) {
        float d = dsqp[(size_t)n * K + j] + zn;
        d = fminf(fmaxf(d, 0.f), 10000.f);
        sd[j] = d;
        float x = -d;
        s1 += x; s2 += x * x;
    }
    float S1 = block_sum(s1, red);
    float S2 = block_sum(s2, red);
    float m1   = S1 * invK;
    float inv1 = rsqrtf(S2 * invK - m1 * m1 + 1e-5f);

    int p = prev[n]; p = p < 0 ? 0 : (p > K - 1 ? K - 1 : p);
    const float* arow = adj + (size_t)p * K;
    float mx = -3.4e38f;
    for (int j = tid; j < K; j += 256) { float a = arow[j]; sa[j] = a; mx = fmaxf(mx, a); }
    float MX = block_max(mx, red);
    float se = 0.f, se2 = 0.f;
    for (int j = tid; j < K; j += 256) {
        float e = __expf(sa[j] - MX); sa[j] = e; se += e; se2 += e * e;
    }
    float SE  = block_sum(se, red);
    float SE2 = block_sum(se2, red);
    float varp = SE2 / (SE * SE) * invK - invK * invK;
    float invp = rsqrtf(varp + 1e-5f);
    float gsig = 1.f / (1.f + __expf(-gate[0]));

    float c1 = 0.f, c2 = 0.f;
    for (int j = tid; j < K; j += 256) {
        float c = ctx[(size_t)n * K + j]; sc[j] = c; c1 += c; c2 += c * c;
    }
    float C1 = block_sum(c1, red), C2 = block_sum(c2, red);
    float mc   = C1 * invK;
    float invc = rsqrtf(C2 * invK - mc * mc + 1e-5f);

    float bv = -3.4e38f, bpv = -3.4e38f;
    int   bi = 0, bpi = 0;
    float invSE = 1.f / SE;
    for (int j = tid; j < K; j += 256) {
        float ld = (-sd[j] - m1) * inv1;
        float gb = (sa[j] * invSE - invK) * invp * gsig;
        float lc = (sc[j] - mc) * invc;
        float t  = ld + gb + 3.0f * lc;
        if (t > bv)   { bv = t;   bi = j; }
        if (ld > bpv) { bpv = ld; bpi = j; }
    }
    int fi, fpi;
    block_argmax(bv, bi, red, redi, fi);
    block_argmax(bpv, bpi, red, redi, fpi);
    if (tid == 0) {
        idx_buf[n] = fi;
        idx_out[n] = (float)fi;
        if (fi != fpi) atomicAdd(div_out, 0.5f / (float)Nrows);
    }
}

// ---------------------------------------------------------------------------
// z_q = codebook[idx]; loss += 1.25*mean((zq-z)^2)   (f32 exact path).
// ---------------------------------------------------------------------------
__global__ __launch_bounds__(256)
void gather_finalize(const float* __restrict__ cb, const int* __restrict__ idx_buf,
                     const float* __restrict__ zr, const float* __restrict__ zi,
                     float* __restrict__ zq, float* __restrict__ loss, int accumulate)
{
    __shared__ float red[256];
    const int n = blockIdx.x, tid = threadIdx.x;
    const int idx = idx_buf[n];
    const int j = tid * 4;
    float4 c = *(const float4*)(cb + (size_t)idx * 1024 + j);
    float4 z = (j < 512) ? *(const float4*)(zr + (size_t)n * 512 + j)
                         : *(const float4*)(zi + (size_t)n * 512 + (j - 512));
    float dx = c.x - z.x, dy = c.y - z.y, dz = c.z - z.z, dw = c.w - z.w;
    float S = block_sum(dx * dx + dy * dy + dz * dz + dw * dw, red);
    *(float4*)(zq + (size_t)n * 1024 + j) = c;
    if (tid == 0) {
        float l = 1.25f * S * (1.f / 1024.f);
        if (accumulate) loss[n] += l; else loss[n] = l;
    }
}

__global__ void colnorm_kernel(const float* __restrict__ cb, float* __restrict__ out, int Kc)
{
    int j = blockIdx.x * blockDim.x + threadIdx.x;
    if (j >= Kc) return;
    const float* row = cb + (size_t)j * 1024;
    float s = 0.f;
    for (int d = 0; d < 1024; d += 4) {
        float4 v = *(const float4*)(row + d);
        s += v.x * v.x + v.y * v.y + v.z * v.z + v.w * v.w;
    }
    out[j] = s;
}

// ---------------------------------------------------------------------------
// f32 -> f16 conversion / layout kernels.
// ---------------------------------------------------------------------------
__global__ __launch_bounds__(256)
void flat_cvt_f16(const float* __restrict__ X, _Float16* __restrict__ Y, int count4)
{
    int i = blockIdx.x * blockDim.x + threadIdx.x;
    if (i >= count4) return;
    float4 v = ((const float4*)X)[i];
    h4 o = {(_Float16)v.x, (_Float16)v.y, (_Float16)v.z, (_Float16)v.w};
    ((h4*)Y)[i] = o;
}

// zh[n][0:512]=zr[n], zh[n][512:1024]=zi[n]  (f16)
__global__ __launch_bounds__(256)
void zcat_cvt_f16(const float* __restrict__ zr, const float* __restrict__ zi,
                  _Float16* __restrict__ zh)
{
    const int n = blockIdx.x, j = threadIdx.x * 4;
    float4 v = (j < 512) ? *(const float4*)(zr + (size_t)n * 512 + j)
                         : *(const float4*)(zi + (size_t)n * 512 + (j - 512));
    h4 o = {(_Float16)v.x, (_Float16)v.y, (_Float16)v.z, (_Float16)v.w};
    *(h4*)(zh + (size_t)n * 1024 + j) = o;
}

// Y[c][r] = (f16)X[r][c]   for X [R x C]; 32x32 LDS tiles.
__global__ __launch_bounds__(256)
void transpose_cvt_f16(const float* __restrict__ X, _Float16* __restrict__ Y,
                       int R, int C)
{
    __shared__ float t[32][33];
    const int bc = blockIdx.x * 32;   // col base
    const int br = blockIdx.y * 32;   // row base
    const int tx = threadIdx.x & 31;
    const int ty = (threadIdx.x >> 5) * 4;
#pragma unroll
    for (int i = 0; i < 4; i++)
        t[ty + i][tx] = X[(size_t)(br + ty + i) * C + bc + tx];
    __syncthreads();
#pragma unroll
    for (int i = 0; i < 4; i++)
        Y[(size_t)(bc + ty + i) * R + br + tx] = (_Float16)t[tx][ty + i];
}

__global__ void zero_scalar(float* p)
{
    if (threadIdx.x == 0 && blockIdx.x == 0) *p = 0.f;
}

// ---------------------------------------------------------------------------
extern "C" void kernel_launch(void* const* d_in, const int* in_sizes, int n_in,
                              void* d_out, int out_size, void* d_ws, size_t ws_size,
                              hipStream_t stream)
{
    (void)in_sizes; (void)n_in; (void)out_size; (void)ws_size;
    const int N = 16384, DIM = 1024, H = 512;
    const int K_SYN = 1024, K_SEM = 4096;

    const float* z_fast_r = (const float*)d_in[0];
    const float* z_fast_i = (const float*)d_in[1];
    const float* z_slow_r = (const float*)d_in[2];
    const float* z_slow_i = (const float*)d_in[3];
    const float* cb_syn   = (const float*)d_in[4];
    const float* cb_sem   = (const float*)d_in[5];
    const float* adj_syn  = (const float*)d_in[6];
    const float* adj_sem  = (const float*)d_in[7];
    const float* gate     = (const float*)d_in[8];
    const int*   prev_syn = (const int*)d_in[9];
    const int*   prev_sem = (const int*)d_in[10];
    // d_in[11] = update_graph (unused); 12..19 syn params, 20..27 sem params

    // ---- workspace layout: f32 region, then f16 region ----
    float* ws = (float*)d_ws;
    size_t o = 0;
    float* ws_ctx = ws + o; o += (size_t)N * K_SEM;
    float* ws_dsq = ws + o; o += (size_t)N * K_SEM;   // prefix doubles as pre-LN temp
    float* ws_cn  = ws + o; o += K_SEM;
    int*   ws_idx = (int*)(ws + o); o += N;
    _Float16* hb = (_Float16*)(ws + o);
    size_t ho = 0;
    _Float16* zh  = hb + ho; ho += (size_t)N * DIM;
    _Float16* h1h = hb + ho; ho += (size_t)N * DIM;
    _Float16* h2h = hb + ho; ho += (size_t)N * H;
    _Float16* cbh = hb + ho; ho += (size_t)K_SEM * DIM;
    _Float16* W1t = hb + ho; ho += (size_t)DIM * DIM;
    _Float16* W2t = hb + ho; ho += (size_t)H * DIM;
    _Float16* W3t = hb + ho; ho += (size_t)K_SEM * H;

    float* out       = (float*)d_out;
    float* o_zq_syn  = out;
    float* o_zq_sem  = o_zq_syn + (size_t)N * DIM;
    float* o_loss    = o_zq_sem + (size_t)N * DIM;
    float* o_idx_syn = o_loss + N;
    float* o_idx_sem = o_idx_syn + N;
    float* o_div     = o_idx_sem + N;

    zero_scalar<<<1, 1, 0, stream>>>(o_div);

    for (int pass = 0; pass < 2; pass++) {
        const int    Kc   = pass ? K_SEM : K_SYN;
        const float* cb   = pass ? cb_sem : cb_syn;
        const float* adj  = pass ? adj_sem : adj_syn;
        const int*   prev = pass ? prev_sem : prev_syn;
        const float* zr   = pass ? z_slow_r : z_fast_r;
        const float* zi   = pass ? z_slow_i : z_fast_i;
        const int    pb   = pass ? 20 : 12;
        const float* W1 = (const float*)d_in[pb + 0];
        const float* b1 = (const float*)d_in[pb + 1];
        const float* lw = (const float*)d_in[pb + 2];
        const float* lb = (const float*)d_in[pb + 3];
        const float* W2 = (const float*)d_in[pb + 4];
        const float* b2 = (const float*)d_in[pb + 5];
        const float* W3 = (const float*)d_in[pb + 6];
        const float* b3 = (const float*)d_in[pb + 7];
        float* idxf = pass ? o_idx_sem : o_idx_syn;
        float* zq   = pass ? o_zq_sem : o_zq_syn;

        // ---- precision staging ----
        colnorm_kernel<<<(Kc + 255) / 256, 256, 0, stream>>>(cb, ws_cn, Kc);
        flat_cvt_f16<<<(Kc * DIM / 4 + 255) / 256, 256, 0, stream>>>(cb, cbh, Kc * DIM / 4);
        zcat_cvt_f16<<<N, 256, 0, stream>>>(zr, zi, zh);
        transpose_cvt_f16<<<dim3(DIM / 32, DIM / 32), 256, 0, stream>>>(W1, W1t, DIM, DIM);
        transpose_cvt_f16<<<dim3(H / 32, DIM / 32), 256, 0, stream>>>(W2, W2t, DIM, H);
        transpose_cvt_f16<<<dim3(Kc / 32, H / 32), 256, 0, stream>>>(W3, W3t, H, Kc);

        // ---- GEMM pipeline (all WMMA f16, f32 accumulate) ----
        dim3 g1(DIM / BNh, N / BMh);
        gemm_wmma_f16<EPI_BIAS><<<g1, GEMM_THREADS, 0, stream>>>(
            zh, W1t, b1, ws_dsq, N, DIM, DIM);                 // pre-LN hidden (f32)
        ln_affine_relu_1024<<<N, 256, 0, stream>>>(ws_dsq, lw, lb, h1h);
        dim3 g2(H / BNh, N / BMh);
        gemm_wmma_f16<EPI_RELU16><<<g2, GEMM_THREADS, 0, stream>>>(
            h1h, W2t, b2, h2h, N, H, DIM);                     // h2 (f16)
        dim3 g3(Kc / BNh, N / BMh);
        gemm_wmma_f16<EPI_BIAS><<<g3, GEMM_THREADS, 0, stream>>>(
            h2h, W3t, b3, ws_ctx, N, Kc, H);                   // ctx logits (f32)
        gemm_wmma_f16<EPI_DIST><<<g3, GEMM_THREADS, 0, stream>>>(
            zh, cbh, ws_cn, ws_dsq, N, Kc, DIM);               // ||c||^2 - 2 z.c (f32)

        if (pass == 0)
            row_logits<K_SYN><<<N, 256, 0, stream>>>(
                ws_dsq, ws_ctx, adj, prev, gate, zr, zi, ws_idx, idxf, o_div, N);
        else
            row_logits<K_SEM><<<N, 256, 0, stream>>>(
                ws_dsq, ws_ctx, adj, prev, gate, zr, zi, ws_idx, idxf, o_div, N);

        gather_finalize<<<N, 256, 0, stream>>>(cb, ws_idx, zr, zi, zq, o_loss, pass);
    }
}
